// SelfAttention_20486994001993
// MI455X (gfx1250) — matrix operand order repaired and checked by reading, not compiled, not run on hardware
//
#include <hip/hip_runtime.h>
#include <hip/hip_bf16.h>

typedef __bf16 bf16_t;
typedef __attribute__((ext_vector_type(16))) __bf16 v16bf;
typedef __attribute__((ext_vector_type(8)))  float  v8f;

union Frag {
    v16bf  v;
    uint4  u[2];
    bf16_t e[16];
};

static __device__ __forceinline__ v8f wmma_bf16(const Frag& a, const Frag& b, v8f c) {
    return __builtin_amdgcn_wmma_f32_16x16x32_bf16(
        /*neg_a=*/false, a.v, /*neg_b=*/false, b.v,
        /*c_mod=*/(short)0, c, /*reuse_a=*/false, /*reuse_b=*/false);
}

// LDS byte-offset of a generic pointer into a __shared__ array (for async-to-LDS vdst)
static __device__ __forceinline__ unsigned lds_offset(const void* p) {
    return (unsigned)(unsigned long long)(__attribute__((address_space(3))) const void*)p;
}

// Async global -> LDS 16-byte copy (CDNA5 Tensor/async path, tracked by ASYNCcnt)
static __device__ __forceinline__ void async_copy_b128(unsigned lds_off, const void* gaddr) {
    asm volatile("global_load_async_to_lds_b128 %0, %1, off"
                 :: "v"(lds_off), "v"((unsigned long long)gaddr)
                 : "memory");
}
static __device__ __forceinline__ void wait_asynccnt0() {
    asm volatile("s_wait_asynccnt 0" ::: "memory");
}

// ---------------------------------------------------------------- convert (8 elems/thread, 16B stores)
__global__ void cvt_f32_to_bf16(const float* __restrict__ in, bf16_t* __restrict__ out, int n) {
    int i = (blockIdx.x * blockDim.x + threadIdx.x) * 8;
    if (i >= n) return;
    float4 a = *reinterpret_cast<const float4*>(in + i);
    float4 b = *reinterpret_cast<const float4*>(in + i + 4);
    union { uint4 u; bf16_t e[8]; } p;
    p.e[0] = (bf16_t)a.x; p.e[1] = (bf16_t)a.y; p.e[2] = (bf16_t)a.z; p.e[3] = (bf16_t)a.w;
    p.e[4] = (bf16_t)b.x; p.e[5] = (bf16_t)b.y; p.e[6] = (bf16_t)b.z; p.e[7] = (bf16_t)b.w;
    *reinterpret_cast<uint4*>(out + i) = p.u;
}

// ---------------------------------------------------------------- GEMM  C[M,N] = A[M,K] * B[N,K]^T
template<bool F32_OUT>
__global__ __launch_bounds__(256)
void gemm_bf16(const bf16_t* __restrict__ A, const bf16_t* __restrict__ B,
               bf16_t* __restrict__ Cb, float* __restrict__ Cf,
               const float* __restrict__ bias, int M, int N, int K) {
    __shared__ __align__(16) bf16_t As[64 * 32];
    __shared__ __align__(16) bf16_t Bs[128 * 32];

    const int tid  = threadIdx.x;
    const int lane = tid & 31;
    const int w    = tid >> 5;
    const int wm   = w >> 2;   // 0..1
    const int wn   = w & 3;    // 0..3
    const int bm   = blockIdx.y;
    const int bn   = blockIdx.x;

    v8f acc[2][2] = {};

    const int ka = (lane < 16) ? 0 : 8;    // A-fragment K base
    const int kb = (lane < 16) ? 0 : 8;    // B-fragment K base

    for (int kc = 0; kc < K; kc += 32) {
        {   // stage A tile 64x32 : one 16B async chunk per thread
            const int row = tid >> 2, seg = tid & 3;
            const bf16_t* src = A + (size_t)(bm * 64 + row) * K + kc + seg * 8;
            async_copy_b128(lds_offset(&As[row * 32 + seg * 8]), src);
            if (kc + 32 < K) __builtin_prefetch(src + 32, 0, 0);
        }
#pragma unroll
        for (int c = 0; c < 2; ++c) {   // stage B tile 128x32
            const int idx = tid * 2 + c;
            const int row = idx >> 2, seg = idx & 3;
            const bf16_t* src = B + (size_t)(bn * 128 + row) * K + kc + seg * 8;
            async_copy_b128(lds_offset(&Bs[row * 32 + seg * 8]), src);
            if (kc + 32 < K) __builtin_prefetch(src + 32, 0, 0);
        }
        wait_asynccnt0();
        __syncthreads();

        Frag af[2], bfrag[2];
#pragma unroll
        for (int i = 0; i < 2; ++i) {
            const int row = wm * 32 + i * 16 + (lane & 15);
            af[i].u[0] = *reinterpret_cast<const uint4*>(&As[row * 32 + ka]);
            af[i].u[1] = *reinterpret_cast<const uint4*>(&As[row * 32 + ka + 16]);
        }
#pragma unroll
        for (int j = 0; j < 2; ++j) {
            const int row = wn * 32 + j * 16 + (lane & 15);
            bfrag[j].u[0] = *reinterpret_cast<const uint4*>(&Bs[row * 32 + kb]);
            bfrag[j].u[1] = *reinterpret_cast<const uint4*>(&Bs[row * 32 + kb + 16]);
        }
#pragma unroll
        for (int i = 0; i < 2; ++i)
#pragma unroll
            for (int j = 0; j < 2; ++j)
                acc[i][j] = wmma_bf16(af[i], bfrag[j], acc[i][j]);
        __syncthreads();
    }

    const int colb   = lane & 15;
    const int rowadd = (lane < 16) ? 0 : 8;
#pragma unroll
    for (int i = 0; i < 2; ++i)
#pragma unroll
        for (int j = 0; j < 2; ++j)
#pragma unroll
            for (int r = 0; r < 8; ++r) {
                const int m = bm * 64 + wm * 32 + i * 16 + r + rowadd;
                const int n = bn * 128 + wn * 32 + j * 16 + colb;
                const float v = acc[i][j][r];
                if (F32_OUT)
                    Cf[(size_t)m * N + n] = v + bias[n];
                else
                    Cb[(size_t)m * N + n] = (bf16_t)v;
            }
}

// ---------------------------------------------------------------- flash attention (S computed transposed)
__global__ __launch_bounds__(128)
void attn_kernel(const bf16_t* __restrict__ Q, const bf16_t* __restrict__ Kmat,
                 const bf16_t* __restrict__ V, bf16_t* __restrict__ O) {
    constexpr int T = 2048, D = 1024, HD = 64;

    __shared__ __align__(16) bf16_t Kc[32 * 64];    // [key][hd]
    __shared__ __align__(16) bf16_t VcT[64 * 32];   // [hd][key]   (transposed)

    const int tid  = threadIdx.x;
    const int lane = tid & 31;
    const int w    = tid >> 5;

    const int bh   = blockIdx.x >> 5;   // 0..31
    const int qblk = blockIdx.x & 31;   // 0..31
    const int b    = bh >> 4;
    const int h    = bh & 15;
    const int q0   = qblk * 64 + w * 16;
    const int colbase = h * HD;

    const int ka  = (lane < 16) ? 0 : 8;    // A-fragment K base
    const int kb  = (lane < 16) ? 0 : 8;    // B-fragment K base
    const int l15 = lane & 15;

    const bf16_t* Kbase = Kmat + (size_t)b * T * D + colbase;
    const bf16_t* Vbase = V    + (size_t)b * T * D + colbase;

    // preload Q as B-fragments of S^T = K * Q^T  (lane = query column)
    Frag qf[2];
    {
        const bf16_t* qptr = Q + ((size_t)b * T + q0 + l15) * D + colbase;
#pragma unroll
        for (int ks = 0; ks < 2; ++ks) {
            qf[ks].u[0] = *reinterpret_cast<const uint4*>(qptr + ks * 32 + kb);
            qf[ks].u[1] = *reinterpret_cast<const uint4*>(qptr + ks * 32 + kb + 16);
        }
    }

    float mrun = -3.0e38f, lrun = 0.0f;   // per-lane scalars (query = lane&15)
    v8f oacc[4] = {};

    for (int jc = 0; jc < T; jc += 32) {
        // ---- K chunk: pure copy -> async global->LDS b128 (ASYNCcnt path)
#pragma unroll
        for (int c = 0; c < 2; ++c) {
            const int idx = tid * 2 + c;        // 0..255
            const int key = idx >> 3, seg = idx & 7;
            async_copy_b128(lds_offset(&Kc[key * 64 + seg * 8]),
                            Kbase + (size_t)(jc + key) * D + seg * 8);
        }
        // ---- V chunk: transpose while staging; pack key-pairs -> ds_store_b32
#pragma unroll
        for (int c = 0; c < 2; ++c) {
            const int idx  = tid * 2 + c;       // 0..255
            const int pair = idx >> 4;          // key pair 0..15
            const int seg  = idx & 15;          // 4 hd elems per seg
            const bf16_t* v0 = Vbase + (size_t)(jc + 2 * pair) * D + seg * 4;
            union { uint2 u; bf16_t e[4]; } r0, r1;
            r0.u = *reinterpret_cast<const uint2*>(v0);
            r1.u = *reinterpret_cast<const uint2*>(v0 + D);
#pragma unroll
            for (int e = 0; e < 4; ++e) {
                union { unsigned u; bf16_t e2[2]; } pk;
                pk.e2[0] = r0.e[e];
                pk.e2[1] = r1.e[e];
                *reinterpret_cast<unsigned*>(&VcT[(seg * 4 + e) * 32 + 2 * pair]) = pk.u;
            }
        }
        wait_asynccnt0();
        __syncthreads();

        // S^T (32 keys x 16 queries) = Kc * Q^T : two accumulators (key tiles 0-15 / 16-31)
        v8f st0 = {}, st1 = {};
#pragma unroll
        for (int ks = 0; ks < 2; ++ks) {
            Frag k0, k1;
            k0.u[0] = *reinterpret_cast<const uint4*>(&Kc[l15 * 64 + ks * 32 + ka]);
            k0.u[1] = *reinterpret_cast<const uint4*>(&Kc[l15 * 64 + ks * 32 + ka + 16]);
            k1.u[0] = *reinterpret_cast<const uint4*>(&Kc[(16 + l15) * 64 + ks * 32 + ka]);
            k1.u[1] = *reinterpret_cast<const uint4*>(&Kc[(16 + l15) * 64 + ks * 32 + ka + 16]);
            st0 = wmma_bf16(k0, qf[ks], st0);
            st1 = wmma_bf16(k1, qf[ks], st1);
        }

        // online softmax: one query per lane (halves paired via xor-16)
        float s0v[8], s1v[8];
        float mx = -3.0e38f;
#pragma unroll
        for (int r = 0; r < 8; ++r) {
            s0v[r] = st0[r] * 0.125f;          // 1/sqrt(64)
            s1v[r] = st1[r] * 0.125f;
            mx = fmaxf(mx, fmaxf(s0v[r], s1v[r]));
        }
        mx = fmaxf(mx, __shfl_xor(mx, 16, 32));
        const float mnew  = fmaxf(mrun, mx);
        const float alpha = __expf(mrun - mnew);
        Frag pf;
        float psum = 0.0f;
#pragma unroll
        for (int r = 0; r < 8; ++r) {
            const float e0 = __expf(s0v[r] - mnew);
            const float e1 = __expf(s1v[r] - mnew);
            psum += e0 + e1;
            pf.e[r]     = (bf16_t)e0;          // in-register C->A fragment conversion
            pf.e[r + 8] = (bf16_t)e1;
        }
        psum += __shfl_xor(psum, 16, 32);
        lrun = lrun * alpha + psum;
        mrun = mnew;

        // rescale O accumulators: element r needs alpha of query (r + 8*(lane>=16))
        const int srcbase = (lane < 16) ? 0 : 8;
#pragma unroll
        for (int r = 0; r < 8; ++r) {
            const float aq = __shfl(alpha, srcbase + r, 32);
#pragma unroll
            for (int j = 0; j < 4; ++j) oacc[j][r] *= aq;
        }

        // O += P * Vc   (A = P fragment in registers, B = VcT rows)
#pragma unroll
        for (int nh = 0; nh < 4; ++nh) {
            Frag vf;
            const int vrow = nh * 16 + l15;
            vf.u[0] = *reinterpret_cast<const uint4*>(&VcT[vrow * 32 + kb]);
            vf.u[1] = *reinterpret_cast<const uint4*>(&VcT[vrow * 32 + kb + 16]);
            oacc[nh] = wmma_bf16(pf, vf, oacc[nh]);
        }
        __syncthreads();
    }

    // epilogue: divide by row sum (broadcast per element), store bf16 O
    const float linv = 1.0f / lrun;
    const int srcbase = (lane < 16) ? 0 : 8;
#pragma unroll
    for (int nh = 0; nh < 4; ++nh)
#pragma unroll
        for (int r = 0; r < 8; ++r) {
            const float lq = __shfl(linv, srcbase + r, 32);
            const int m = q0 + r + ((lane < 16) ? 0 : 8);
            const int n = colbase + nh * 16 + l15;
            O[((size_t)b * T + m) * D + n] = (bf16_t)(oacc[nh][r] * lq);
        }
}

// ---------------------------------------------------------------- launch
extern "C" void kernel_launch(void* const* d_in, const int* in_sizes, int n_in,
                              void* d_out, int out_size, void* d_ws, size_t ws_size,
                              hipStream_t stream) {
    (void)in_sizes; (void)n_in; (void)out_size; (void)ws_size;
    const float* x  = (const float*)d_in[0];
    const float* Wq = (const float*)d_in[1];
    const float* Wk = (const float*)d_in[2];
    const float* Wv = (const float*)d_in[3];
    const float* Wo = (const float*)d_in[4];
    const float* bo = (const float*)d_in[5];

    constexpr int BT = 2 * 2048;      // 4096 rows
    constexpr int Kd = 1024;
    const int NX = BT * Kd;           // 4M elems
    const int NW = Kd * Kd;           // 1M elems

    char* ws = (char*)d_ws;
    bf16_t* xb  = (bf16_t*)(ws);
    bf16_t* Wqb = (bf16_t*)(ws + (size_t)8  * 1024 * 1024);
    bf16_t* Wkb = (bf16_t*)(ws + (size_t)10 * 1024 * 1024);
    bf16_t* Wvb = (bf16_t*)(ws + (size_t)12 * 1024 * 1024);
    bf16_t* Wob = (bf16_t*)(ws + (size_t)14 * 1024 * 1024);
    bf16_t* Qb  = (bf16_t*)(ws + (size_t)16 * 1024 * 1024);
    bf16_t* Kb  = (bf16_t*)(ws + (size_t)24 * 1024 * 1024);
    bf16_t* Vb  = (bf16_t*)(ws + (size_t)32 * 1024 * 1024);
    bf16_t* Ob  = (bf16_t*)(ws + (size_t)40 * 1024 * 1024);

    cvt_f32_to_bf16<<<(NX / 8 + 255) / 256, 256, 0, stream>>>(x,  xb,  NX);
    cvt_f32_to_bf16<<<(NW / 8 + 255) / 256, 256, 0, stream>>>(Wq, Wqb, NW);
    cvt_f32_to_bf16<<<(NW / 8 + 255) / 256, 256, 0, stream>>>(Wk, Wkb, NW);
    cvt_f32_to_bf16<<<(NW / 8 + 255) / 256, 256, 0, stream>>>(Wv, Wvb, NW);
    cvt_f32_to_bf16<<<(NW / 8 + 255) / 256, 256, 0, stream>>>(Wo, Wob, NW);

    dim3 ggrid(Kd / 128, BT / 64);    // (8, 64)
    gemm_bf16<false><<<ggrid, 256, 0, stream>>>(xb, Wqb, Qb, nullptr, nullptr, BT, Kd, Kd);
    gemm_bf16<false><<<ggrid, 256, 0, stream>>>(xb, Wkb, Kb, nullptr, nullptr, BT, Kd, Kd);
    gemm_bf16<false><<<ggrid, 256, 0, stream>>>(xb, Wvb, Vb, nullptr, nullptr, BT, Kd, Kd);

    attn_kernel<<<1024, 128, 0, stream>>>(Qb, Kb, Vb, Ob);

    gemm_bf16<true><<<ggrid, 256, 0, stream>>>(Ob, Wob, nullptr, (float*)d_out, bo, BT, Kd, Kd);
}
